// BuildLstmUnrollNet_44006234914903
// MI455X (gfx1250) — matrix operand order repaired and checked by reading, not compiled, not run on hardware
//
#include <hip/hip_runtime.h>

// Problem constants (from reference): B=256, I=512, R=1024, L=2, U=11
#define BDIM 256
#define IDIM 512
#define RDIM 1024
#define UDIM 11

// GEMM staging geometry (bf16 in LDS)
#define KC 64                        // K-chunk = 2 WMMA k-steps
#define WSTRIDE 72                   // bf16 per staged row (64 + 8 pad -> 144B stride)
#define WROWS 128                    // 4 gates * 32 N-columns per block
#define AROWS 64                     // block M-tile rows
#define WOFF_H 0
#define AOFF_H (WROWS * WSTRIDE)                 // 9216 bf16
#define BUF_H  ((WROWS + AROWS) * WSTRIDE)       // 13824 bf16 = 27,648 B per buffer

typedef __bf16 bf16_t;
typedef __attribute__((ext_vector_type(16))) __bf16 v16bf;
typedef __attribute__((ext_vector_type(8)))  float  v8f;

#define LOG2E  1.4426950408889634f
#define LOG2E2 2.8853900817779268f

__device__ __forceinline__ float fast_sigmoid(float x) {
    return __builtin_amdgcn_rcpf(1.0f + __builtin_amdgcn_exp2f(-LOG2E * x));
}
__device__ __forceinline__ float fast_tanh(float x) {
    float e = __builtin_amdgcn_exp2f(LOG2E2 * x);
    return 1.0f - 2.0f * __builtin_amdgcn_rcpf(e + 1.0f);
}
__device__ __forceinline__ float4 ldg4(const float* p) {
    return *reinterpret_cast<const float4*>(p);
}

// Convert 8 fp32 (two float4) to 8 bf16 and store 16B to LDS.
__device__ __forceinline__ void store_pk8(bf16_t* lp, float4 a, float4 b) {
    union { bf16_t h[8]; uint4 u; } t;
    t.h[0] = (bf16_t)a.x; t.h[1] = (bf16_t)a.y; t.h[2] = (bf16_t)a.z; t.h[3] = (bf16_t)a.w;
    t.h[4] = (bf16_t)b.x; t.h[5] = (bf16_t)b.y; t.h[6] = (bf16_t)b.z; t.h[7] = (bf16_t)b.w;
    *reinterpret_cast<uint4*>(lp) = t.u;
}

// Two 16B LDS reads -> one v16bf fragment (offset of second chunk in bf16 elements).
__device__ __forceinline__ v16bf ld_frag(const bf16_t* p, int off2) {
    union { v16bf v; uint4 q[2]; } u;
    u.q[0] = *reinterpret_cast<const uint4*>(p);
    u.q[1] = *reinterpret_cast<const uint4*>(p + off2);
    return u.v;
}

// ---- cooperative chunk copy, split into load phase (overlaps compute) + store phase ----
// Per thread: W tile 128 rows x 64 K -> 4 row-groups, A tile 64 rows x 64 K -> 2 row-groups.
// Thread t: lo8 = t&7 selects an 8-float K-segment, rgrp = t>>3 selects row in each group.
struct CopyRegs { float4 w[8]; float4 a[4]; };

__device__ __forceinline__ void copy_load(
    const float* __restrict__ W, const float* __restrict__ Act, int K,
    int blockM, int blockN, int kc, int tid, CopyRegs& cr)
{
    const int lo8  = (tid & 7) * 8;     // K offset (floats)
    const int rgrp = tid >> 3;          // 0..31
#pragma unroll
    for (int it = 0; it < 4; ++it) {
        const int r = it * 32 + rgrp;               // staged W row 0..127
        const int g = r >> 5;
        const int n = r & 31;
        const float* gp = W + (long)(g * RDIM + blockN + n) * K + kc + lo8;
        cr.w[2 * it]     = ldg4(gp);
        cr.w[2 * it + 1] = ldg4(gp + 4);
    }
#pragma unroll
    for (int it = 0; it < 2; ++it) {
        const int r = it * 32 + rgrp;               // staged A row 0..63
        const float* gp = Act + (long)(blockM + r) * K + kc + lo8;
        cr.a[2 * it]     = ldg4(gp);
        cr.a[2 * it + 1] = ldg4(gp + 4);
    }
}

__device__ __forceinline__ void copy_store(bf16_t* buf, int tid, const CopyRegs& cr)
{
    const int lo8  = (tid & 7) * 8;
    const int rgrp = tid >> 3;
#pragma unroll
    for (int it = 0; it < 4; ++it) {
        const int r = it * 32 + rgrp;
        store_pk8(buf + WOFF_H + r * WSTRIDE + lo8, cr.w[2 * it], cr.w[2 * it + 1]);
    }
#pragma unroll
    for (int it = 0; it < 2; ++it) {
        const int r = it * 32 + rgrp;
        store_pk8(buf + AOFF_H + r * WSTRIDE + lo8, cr.a[2 * it], cr.a[2 * it + 1]);
    }
}

// Double-buffered, bf16-LDS-staged accumulation of acc[g] += Act @ Wg^T
// for this wave's 16x16 tile (all 4 gates).
__device__ __forceinline__ void gemm_part(
    const float* __restrict__ Act, const float* __restrict__ W, int K,
    int blockM, int blockN, int msel, int nsel,
    v8f acc[4], int lane, int tid, bf16_t* smem)
{
    const int half = lane >> 4;     // half-wave selects K sub-pattern
    const int l15  = lane & 15;
    const int nchunks = K / KC;

    CopyRegs cr;
    copy_load(W, Act, K, blockM, blockN, 0, tid, cr);
    copy_store(smem, tid, cr);
    __syncthreads();

    const bf16_t* abase = smem + AOFF_H + (msel * 16 + l15) * WSTRIDE + half * 8;
    const bf16_t* wbase = smem + WOFF_H + (nsel * 16 + l15) * WSTRIDE + half * 8;

    for (int i = 0; i < nchunks; ++i) {
        const int sel = (i & 1) * BUF_H;
        const bool has_next = (i + 1 < nchunks);
        if (has_next)   // global loads fly while this chunk's ds/wmma work runs
            copy_load(W, Act, K, blockM, blockN, (i + 1) * KC, tid, cr);

#pragma unroll
        for (int ks = 0; ks < 2; ++ks) {            // two 32-deep WMMA steps
            // A fragment: elems 0..7 <- K kb..kb+7, elems 8..15 <- K kb+16..kb+23
            v16bf af = ld_frag(abase + sel + ks * 32, 16);
            v16bf bfr[4];
#pragma unroll
            for (int g = 0; g < 4; ++g)             // preload all 4 B fragments
                bfr[g] = ld_frag(wbase + sel + g * 32 * WSTRIDE + ks * 32, 16);
#pragma unroll
            for (int g = 0; g < 4; ++g)             // 4 back-to-back WMMAs
                acc[g] = __builtin_amdgcn_wmma_f32_16x16x32_bf16(
                    false, af, false, bfr[g], (short)0, acc[g], false, false);
        }

        __syncthreads();                            // all waves done reading
        if (has_next) {
            copy_store(smem + ((i + 1) & 1) * BUF_H, tid, cr);
            __syncthreads();                        // staged data visible
        }
    }
}

// Fused LSTM layer step:
//   pre = A1 @ W1^T + b1 + A2 @ W2^T + b2   (4R gate columns)
//   c_out = sigm(pre_f)*c_prev + sigm(pre_i)*tanh(pre_g);  h_out = sigm(pre_o)*tanh(c_out)
// Block = 8 waves: 64(M) x 32(N) h-patch; each wave owns 16x16 with all 4 gates.
__global__ __launch_bounds__(256) void lstm_layer_kernel(
    const float* __restrict__ A1, const float* __restrict__ W1, int K1,
    const float* __restrict__ A2, const float* __restrict__ W2, int K2,
    const float* __restrict__ b1, const float* __restrict__ b2,
    const float* __restrict__ c_prev,
    float* __restrict__ h_out, float* __restrict__ c_out)
{
    __shared__ __align__(16) bf16_t smem[2 * BUF_H];   // 55,296 B

    const int tid    = threadIdx.x;
    const int lane   = tid & 31;
    const int wave   = tid >> 5;
    const int msel   = wave & 3;
    const int nsel   = wave >> 2;
    const int blockM = blockIdx.x * 64;
    const int blockN = blockIdx.y * 32;
    const int l15    = lane & 15;
    const int ncol   = blockN + nsel * 16 + l15;       // column within R

    v8f acc[4];
#pragma unroll
    for (int g = 0; g < 4; ++g) {
        const float bias = b1[g * RDIM + ncol] + b2[g * RDIM + ncol];
#pragma unroll
        for (int e = 0; e < 8; ++e) acc[g][e] = bias;
    }

    gemm_part(A1, W1, K1, blockM, blockN, msel, nsel, acc, lane, tid, smem);
    gemm_part(A2, W2, K2, blockM, blockN, msel, nsel, acc, lane, tid, smem);

    // C/D layout: lane<16 -> M = base+e ; lane>=16 -> M = base+e+8 ; N = lane&15
    const int mrow0 = blockM + msel * 16 + ((lane & 16) ? 8 : 0);
#pragma unroll
    for (int e = 0; e < 8; ++e) {
        const long idx = (long)(mrow0 + e) * RDIM + ncol;
        const float cp = c_prev[idx];
        const float gi = fast_sigmoid(acc[0][e]);
        const float gf = fast_sigmoid(acc[1][e]);
        const float go = fast_sigmoid(acc[2][e]);
        const float gg = fast_tanh(acc[3][e]);
        const float c  = gf * cp + gi * gg;
        const float h  = go * fast_tanh(c);
        c_out[idx] = c;
        h_out[idx] = h;
    }
}

// Unpack init_states_input (B, 4, R) -> h0, c0, h1, c1.
__global__ void init_states_kernel(const float* __restrict__ st,
                                   float* __restrict__ h0, float* __restrict__ c0,
                                   float* __restrict__ h1, float* __restrict__ c1)
{
    const int  m    = blockIdx.x;
    const long base = (long)m * 4 * RDIM;
    const long row  = (long)m * RDIM;
    for (int n = threadIdx.x; n < RDIM; n += blockDim.x) {
        h0[row + n] = st[base + 0 * RDIM + n];
        c0[row + n] = st[base + 1 * RDIM + n];
        h1[row + n] = st[base + 2 * RDIM + n];
        c1[row + n] = st[base + 3 * RDIM + n];
    }
}

// ys[t] = h1 (pre-update): out[m, t*R + n] = h1[m, n]
__global__ void copy_out_kernel(const float* __restrict__ h1,
                                float* __restrict__ out, int t)
{
    const int m = blockIdx.x;
    const float4* src = reinterpret_cast<const float4*>(h1 + (long)m * RDIM);
    float4* dst = reinterpret_cast<float4*>(out + (long)m * (UDIM * RDIM) + (long)t * RDIM);
    for (int n = threadIdx.x; n < RDIM / 4; n += blockDim.x) dst[n] = src[n];
}

extern "C" void kernel_launch(void* const* d_in, const int* in_sizes, int n_in,
                              void* d_out, int out_size, void* d_ws, size_t ws_size,
                              hipStream_t stream)
{
    const float* x   = (const float*)d_in[0];   // (B, I)
    const float* st  = (const float*)d_in[1];   // (B, 2L*R)
    const float* Wi0 = (const float*)d_in[2];   // (U, 4R, I)
    const float* bi0 = (const float*)d_in[3];   // (U, 4R)
    const float* Wh0 = (const float*)d_in[4];   // (U, 4R, R)
    const float* bh0 = (const float*)d_in[5];   // (U, 4R)
    const float* Wi1 = (const float*)d_in[6];   // (U, 4R, R)
    const float* bi1 = (const float*)d_in[7];   // (U, 4R)
    const float* Wh1 = (const float*)d_in[8];   // (U, 4R, R)
    const float* bh1 = (const float*)d_in[9];   // (U, 4R)
    float* out = (float*)d_out;                 // (B, U*R)
    float* ws  = (float*)d_ws;

    const long S = (long)BDIM * RDIM;
    float* h0a = ws + 0 * S;  float* h0b = ws + 1 * S;
    float* c0a = ws + 2 * S;  float* c0b = ws + 3 * S;
    float* h1a = ws + 4 * S;  float* h1b = ws + 5 * S;
    float* c1a = ws + 6 * S;  float* c1b = ws + 7 * S;

    init_states_kernel<<<BDIM, 256, 0, stream>>>(st, h0a, c0a, h1a, c1a);

    float *h0c = h0a, *h0n = h0b, *c0c = c0a, *c0n = c0b;
    float *h1c = h1a, *h1n = h1b, *c1c = c1a, *c1n = c1b;

    const dim3 grid(BDIM / 64, RDIM / 32);      // (4, 32) = 128 blocks
    const long wsz0 = 4L * RDIM * IDIM;
    const long wszh = 4L * RDIM * RDIM;
    const long bsz  = 4L * RDIM;

    for (int t = 0; t < UDIM; ++t) {
        copy_out_kernel<<<BDIM, 256, 0, stream>>>(h1c, out, t);

        lstm_layer_kernel<<<grid, 256, 0, stream>>>(
            x,   Wi0 + (long)t * wsz0, IDIM,
            h0c, Wh0 + (long)t * wszh, RDIM,
            bi0 + (long)t * bsz, bh0 + (long)t * bsz,
            c0c, h0n, c0n);

        lstm_layer_kernel<<<grid, 256, 0, stream>>>(
            h0n, Wi1 + (long)t * wszh, RDIM,
            h1c, Wh1 + (long)t * wszh, RDIM,
            bi1 + (long)t * bsz, bh1 + (long)t * bsz,
            c1c, h1n, c1n);

        float* tmp;
        tmp = h0c; h0c = h0n; h0n = tmp;
        tmp = c0c; c0c = c0n; c0n = tmp;
        tmp = h1c; h1c = h1n; h1n = tmp;
        tmp = c1c; c1c = c1n; c1n = tmp;
    }
}
